// MobileEMA_29102698397816
// MI455X (gfx1250) — matrix operand order repaired and checked by reading, not compiled, not run on hardware
//
#include <hip/hip_runtime.h>
#include <hip/hip_bf16.h>
#include <cmath>

// Problem constants (from reference): x (16,256,64,64) fp32, GROUPS=32, gc=8
#define BATCH 16
#define CHN   256
#define HWSZ  4096      // 64*64
#define NG    32
#define GC    8
#define BG    512       // BATCH*NG

typedef __attribute__((ext_vector_type(16))) __bf16 v16bf;
typedef __attribute__((ext_vector_type(4)))  __bf16 v4bf;
typedef __attribute__((ext_vector_type(8)))  float  v8f;
typedef __attribute__((ext_vector_type(4)))  int    v4i;

__device__ __forceinline__ float sigmoidf(float x) { return 1.0f / (1.0f + __expf(-x)); }

// ---------------------------------------------------------------------------
// CDNA5 async global->LDS copy (ASYNCcnt path). Probe-verified present on this
// toolchain; signature wants int4* payload pointers (AS1 src, AS3 dst).
// ---------------------------------------------------------------------------
#if defined(__gfx1250__) && \
    __has_builtin(__builtin_amdgcn_global_load_async_to_lds_b128) && \
    __has_builtin(__builtin_amdgcn_s_wait_asynccnt)
#define HAVE_ASYNC_LDS 1
#else
#define HAVE_ASYNC_LDS 0
#endif

__device__ __forceinline__ void async_copy16(const float* g, float* l) {
#if HAVE_ASYNC_LDS
  __builtin_amdgcn_global_load_async_to_lds_b128(
      (__attribute__((address_space(1))) v4i*)const_cast<float*>(g),
      (__attribute__((address_space(3))) v4i*)l, 0, 0);
#else
  *(float4*)l = *(const float4*)g;
#endif
}
__device__ __forceinline__ void async_join() {
#if HAVE_ASYNC_LDS
  __builtin_amdgcn_s_wait_asynccnt(0);
#endif
}

// ---------------------------------------------------------------------------
// K1: per-(b,c) mean+max over spatial -> svec = avg + mx   (input to catt MLP)
// ---------------------------------------------------------------------------
__global__ __launch_bounds__(256) void k_stats(const float* __restrict__ x,
                                               float* __restrict__ svec) {
  int c = blockIdx.x, b = blockIdx.y, tid = threadIdx.x;
  const float* row = x + ((size_t)b * CHN + c) * HWSZ;
  float s = 0.f, m = -INFINITY;
  for (int i = tid; i < HWSZ; i += 256) { float v = row[i]; s += v; m = fmaxf(m, v); }
  __shared__ float ss[256], sm[256];
  ss[tid] = s; sm[tid] = m; __syncthreads();
  for (int off = 128; off > 0; off >>= 1) {
    if (tid < off) { ss[tid] += ss[tid + off]; sm[tid] = fmaxf(sm[tid], sm[tid + off]); }
    __syncthreads();
  }
  if (tid == 0) svec[b * CHN + c] = ss[0] * (1.0f / HWSZ) + sm[0];
}

// ---------------------------------------------------------------------------
// K2: catt = sigmoid(relu(s @ ci_w1) @ ci_w2)   (tiny: 16x256x64 — VALU)
// ---------------------------------------------------------------------------
__global__ __launch_bounds__(256) void k_catt(const float* __restrict__ svec,
                                              const float* __restrict__ w1,
                                              const float* __restrict__ w2,
                                              float* __restrict__ catt) {
  int b = blockIdx.x, tid = threadIdx.x;
  __shared__ float srow[CHN];
  __shared__ float hb[64];
  srow[tid] = svec[b * CHN + tid];
  __syncthreads();
  if (tid < 64) {
    float a = 0.f;
    for (int k = 0; k < CHN; k++) a += srow[k] * w1[k * 64 + tid];
    hb[tid] = fmaxf(a, 0.f);
  }
  __syncthreads();
  float a = 0.f;
  for (int j = 0; j < 64; j++) a += hb[j] * w2[j * CHN + tid];
  catt[b * CHN + tid] = sigmoidf(a);
}

// ---------------------------------------------------------------------------
// K3: cex = x + x*catt + ci*(1-catt), ci = grouped 1x1 conv (32 groups of 8)
//     Block: one group (8 ch) x 256 spatial positions; x staged once in LDS
//     via async global->LDS copies (8 KB verbatim).
// ---------------------------------------------------------------------------
__global__ __launch_bounds__(256) void k_cex(const float* __restrict__ x,
                                             const float* __restrict__ catt,
                                             const float* __restrict__ wci,
                                             float* __restrict__ cex) {
  int nb = blockIdx.x, g = blockIdx.y, b = blockIdx.z, tid = threadIdx.x;
  __shared__ float xs[8][256];
  size_t base = ((size_t)b * CHN + g * 8) * HWSZ + nb * 256;
  // 8 rows x 256 floats; 512 x 16B packets, 2 per thread
  for (int q = tid; q < 512; q += 256) {
    int k = q >> 6;
    int col = (q & 63) * 4;
    async_copy16(x + base + (size_t)k * HWSZ + col, &xs[k][col]);
  }
  async_join();
  __syncthreads();
  for (int ch = 0; ch < 8; ch++) {
    int c = g * 8 + ch;
    float ci = 0.f;
    for (int k = 0; k < 8; k++) ci += wci[c * 8 + k] * xs[k][tid];
    float a = catt[b * CHN + c];
    float xv = xs[ch][tid];
    cex[base + (size_t)ch * HWSZ + tid] = xv + xv * a + ci * (1.f - a);
  }
}

// ---------------------------------------------------------------------------
// K4: per-(bg,ch) row means (over w) and column means (over h) of gx(=cex)
// ---------------------------------------------------------------------------
__global__ __launch_bounds__(64) void k_rowcol(const float* __restrict__ gx,
                                               float* __restrict__ mh,
                                               float* __restrict__ mw) {
  int ch = blockIdx.x, bg = blockIdx.y, t = threadIdx.x; // 64 threads
  const float* tile = gx + ((size_t)bg * GC + ch) * HWSZ;
  float rs = 0.f, cs = 0.f;
  for (int j = 0; j < 64; j++) { rs += tile[t * 64 + j]; cs += tile[j * 64 + t]; }
  mh[((size_t)bg * GC + ch) * 64 + t] = rs * (1.f / 64.f);
  mw[((size_t)bg * GC + ch) * 64 + t] = cs * (1.f / 64.f);
}

// ---------------------------------------------------------------------------
// K5: 1x1 conv over 8 ch on concat([x_h; x_w], axis=2), store sigmoids
// ---------------------------------------------------------------------------
__global__ __launch_bounds__(128) void k_ema1(const float* __restrict__ mh,
                                              const float* __restrict__ mw,
                                              const float* __restrict__ w1,
                                              float* __restrict__ xhs,
                                              float* __restrict__ xws) {
  int bg = blockIdx.x, p = threadIdx.x; // 0..127
  __shared__ float in[8][128];
  for (int k = 0; k < 8; k++)
    in[k][p] = (p < 64) ? mh[((size_t)bg * GC + k) * 64 + p]
                        : mw[((size_t)bg * GC + k) * 64 + (p - 64)];
  __syncthreads();
  for (int o = 0; o < 8; o++) {
    float v = 0.f;
    for (int k = 0; k < 8; k++) v += w1[o * 8 + k] * in[k][p];
    float s = sigmoidf(v);
    if (p < 64) xhs[((size_t)bg * GC + o) * 64 + p] = s;
    else        xws[((size_t)bg * GC + o) * 64 + (p - 64)] = s;
  }
}

// ---------------------------------------------------------------------------
// K6: x1 = per_channel_gn(gx * sig(xh)_row * sig(xw)_col)  (t staged in LDS)
// ---------------------------------------------------------------------------
__global__ __launch_bounds__(256) void k_x1gn(const float* __restrict__ gx,
                                              const float* __restrict__ xhs,
                                              const float* __restrict__ xws,
                                              const float* __restrict__ gn_g,
                                              const float* __restrict__ gn_b,
                                              float* __restrict__ x1) {
  int ch = blockIdx.x, bg = blockIdx.y, tid = threadIdx.x;
  __shared__ float tbuf[HWSZ];
  __shared__ float rs[256], rq[256];
  size_t base = ((size_t)bg * GC + ch) * HWSZ;
  size_t vb = ((size_t)bg * GC + ch) * 64;
  float s = 0.f, q = 0.f;
  for (int it = 0; it < 16; it++) {
    int n = it * 256 + tid;
    float v = gx[base + n] * xhs[vb + (n >> 6)] * xws[vb + (n & 63)];
    tbuf[n] = v; s += v; q += v * v;
  }
  rs[tid] = s; rq[tid] = q; __syncthreads();
  for (int off = 128; off > 0; off >>= 1) {
    if (tid < off) { rs[tid] += rs[tid + off]; rq[tid] += rq[tid + off]; }
    __syncthreads();
  }
  float mu  = rs[0] * (1.f / HWSZ);
  float var = rq[0] * (1.f / HWSZ) - mu * mu;
  float inv = rsqrtf(var + 1e-5f);
  float gg = gn_g[ch] * inv;
  float bb = gn_b[ch] - mu * gg;
  for (int it = 0; it < 16; it++) {
    int n = it * 256 + tid;
    x1[base + n] = tbuf[n] * gg + bb;
  }
}

// ---------------------------------------------------------------------------
// K7: x2 = grouped 3x3 conv of gx (4 groups of 2 in-ch), pad=1
//     Block: one (bg, out-ch); both input channels (32 KB, contiguous in
//     global) staged in LDS via async global->LDS copies.
// ---------------------------------------------------------------------------
__global__ __launch_bounds__(256) void k_x2conv(const float* __restrict__ gx,
                                                const float* __restrict__ w3,
                                                float* __restrict__ x2) {
  int och = blockIdx.x, bg = blockIdx.y, tid = threadIdx.x;
  int icb = (och >> 1) << 1;
  __shared__ float zin[2][HWSZ];
  size_t gbase = ((size_t)bg * GC + icb) * HWSZ;
  float* zflat = &zin[0][0];
  for (int idx = tid * 4; idx < 2 * HWSZ; idx += 1024)   // 8 x b128 per thread
    async_copy16(gx + gbase + idx, zflat + idx);
  async_join();
  __syncthreads();
  const float* wp = w3 + och * 18; // (8,2,3,3)
  for (int it = 0; it < 16; it++) {
    int n = it * 256 + tid;
    int i = n >> 6, j = n & 63;
    float acc = 0.f;
    for (int k = 0; k < 2; k++)
      for (int r = 0; r < 3; r++) {
        int ii = i + r - 1; if (ii < 0 || ii >= 64) continue;
        for (int sidx = 0; sidx < 3; sidx++) {
          int jj = j + sidx - 1; if (jj < 0 || jj >= 64) continue;
          acc += wp[k * 9 + r * 3 + sidx] * zin[k][ii * 64 + jj];
        }
      }
    x2[((size_t)bg * GC + och) * HWSZ + n] = acc;
  }
}

// ---------------------------------------------------------------------------
// K8: cross_spatial(z) -> ze. Block = (row-chunk of 16, bg); all 8 channels
//     of an 18-row halo slab in LDS (36 KB). f1 (1x1/8), f3 (3x3 grouped/4),
//     sa = sigmoid(bn(att 1x1/16)), cs (1x1 grouped/2).
// ---------------------------------------------------------------------------
__global__ __launch_bounds__(256) void k_cross(const float* __restrict__ z,
                                               float* __restrict__ ze,
                                               const float* __restrict__ w1,
                                               const float* __restrict__ w3,
                                               const float* __restrict__ wa,
                                               const float* __restrict__ bng,
                                               const float* __restrict__ bnb,
                                               const float* __restrict__ wc) {
  int rc = blockIdx.x, bg = blockIdx.y, tid = threadIdx.x;
  int i0 = rc * 16;
  __shared__ float zt[GC][18 * 64];
  size_t base = (size_t)bg * GC * HWSZ;
  for (int idx = tid; idx < GC * 18 * 64; idx += 256) {
    int k = idx / (18 * 64);
    int rem = idx % (18 * 64);
    int gi = i0 - 1 + (rem >> 6);
    float v = 0.f;
    if (gi >= 0 && gi < 64) v = z[base + (size_t)k * HWSZ + gi * 64 + (rem & 63)];
    zt[k][rem] = v;
  }
  __syncthreads();
  const float bnrs = rsqrtf(1.f + 1e-5f);
  for (int it = 0; it < 4; it++) {
    int n = it * 256 + tid;        // 0..1023 within the 16x64 chunk
    int li = n >> 6, j = n & 63;
    int lr = li + 1;               // LDS row (halo offset)
    float f1[8], f3[8];
    for (int o = 0; o < 8; o++) {
      float a = 0.f;
      for (int k = 0; k < 8; k++) a += w1[o * 8 + k] * zt[k][lr * 64 + j];
      f1[o] = a;
      float b3 = 0.f;
      int icb = (o >> 2) << 2;
      for (int k = 0; k < 4; k++)
        for (int r = 0; r < 3; r++) {
          int rr = lr + r - 1;     // always within 0..17 (halo)
          for (int sidx = 0; sidx < 3; sidx++) {
            int jj = j + sidx - 1; if (jj < 0 || jj >= 64) continue;
            b3 += w3[o * 36 + k * 9 + r * 3 + sidx] * zt[icb + k][rr * 64 + jj];
          }
        }
      f3[o] = b3;
    }
    int gi = i0 + li;
    for (int o = 0; o < 8; o++) {
      float a = 0.f;
      for (int k = 0; k < 8; k++) a += wa[o * 16 + k] * f1[k] + wa[o * 16 + 8 + k] * f3[k];
      float sa = sigmoidf(a * bng[o] * bnrs + bnb[o]);
      int icb = (o >> 1) << 1;
      float cs = wc[o * 2] * zt[icb][lr * 64 + j] + wc[o * 2 + 1] * zt[icb + 1][lr * 64 + j];
      float zv = zt[o][lr * 64 + j];
      ze[base + (size_t)o * HWSZ + gi * 64 + j] = zv + zv * sa + cs * (1.f - sa);
    }
  }
}

// ---------------------------------------------------------------------------
// K9: per-(bg,ch) spatial means of x1e and x2e (softmax inputs)
// ---------------------------------------------------------------------------
__global__ __launch_bounds__(256) void k_pool(const float* __restrict__ x1e,
                                              const float* __restrict__ x2e,
                                              float* __restrict__ m1,
                                              float* __restrict__ m2) {
  int ch = blockIdx.x, bg = blockIdx.y, tid = threadIdx.x;
  size_t base = ((size_t)bg * GC + ch) * HWSZ;
  float a = 0.f, b = 0.f;
  for (int i = tid; i < HWSZ; i += 256) { a += x1e[base + i]; b += x2e[base + i]; }
  __shared__ float ra[256], rb[256];
  ra[tid] = a; rb[tid] = b; __syncthreads();
  for (int off = 128; off > 0; off >>= 1) {
    if (tid < off) { ra[tid] += ra[tid + off]; rb[tid] += rb[tid + off]; }
    __syncthreads();
  }
  if (tid == 0) { m1[bg * GC + ch] = ra[0] * (1.f / HWSZ); m2[bg * GC + ch] = rb[0] * (1.f / HWSZ); }
}

// ---------------------------------------------------------------------------
// K10: softmax(8) x2 -> per-pixel weights = sigmoid(y1+y2); se = gx * weights
// ---------------------------------------------------------------------------
__global__ __launch_bounds__(256) void k_se(const float* __restrict__ m1,
                                            const float* __restrict__ m2,
                                            const float* __restrict__ x1e,
                                            const float* __restrict__ x2e,
                                            const float* __restrict__ gx,
                                            float* __restrict__ se) {
  int bg = blockIdx.x, tid = threadIdx.x;
  __shared__ float s1[8], s2[8];
  if (tid == 0) {
    float mx1 = -INFINITY, mx2 = -INFINITY;
    for (int c = 0; c < 8; c++) {
      mx1 = fmaxf(mx1, m1[bg * GC + c]);
      mx2 = fmaxf(mx2, m2[bg * GC + c]);
    }
    float t1 = 0.f, t2 = 0.f;
    for (int c = 0; c < 8; c++) {
      s1[c] = __expf(m1[bg * GC + c] - mx1); t1 += s1[c];
      s2[c] = __expf(m2[bg * GC + c] - mx2); t2 += s2[c];
    }
    for (int c = 0; c < 8; c++) { s1[c] /= t1; s2[c] /= t2; }
  }
  __syncthreads();
  size_t base = (size_t)bg * GC * HWSZ;
  for (int it = 0; it < 16; it++) {
    int n = it * 256 + tid;
    float y = 0.f;
    for (int c = 0; c < 8; c++)
      y += s1[c] * x2e[base + (size_t)c * HWSZ + n] + s2[c] * x1e[base + (size_t)c * HWSZ + n];
    float wv = sigmoidf(y);
    for (int c = 0; c < 8; c++)
      se[base + (size_t)c * HWSZ + n] = gx[base + (size_t)c * HWSZ + n] * wv;
  }
}

// ---------------------------------------------------------------------------
// K11: batched 1x1-conv GEMM with v_wmma_f32_16x16x32_bf16 + fused BN epilogue.
//   Y[b, m, n] = act( bn( sum_k W[m,k] * X[b,k,n] ) )
//   Block = 8 waves; wave w owns N-tile n0 = bx*128 + w*16 and ALL M-tiles.
//   Weight tiles are double-buffered bf16 in LDS (software pipeline, one
//   barrier per K-step); staging is vectorized: global_load_b128 ->
//   2x v_cvt_pk_bf16_f32 -> ds_store_b64.
//   A k order used here (contiguous, same as B): lanes 0-15 K {0..15}, lanes 16-31 K
//   {16..31}. B: lane<16 col n0+lane K k0..k0+15; lane>=16 col -16, K+16.
//   ACT: 0 = none, 1 = relu, 2 = final: fw=sigmoid(v); out=cex*fw+se*(1-fw).
// ---------------------------------------------------------------------------
template <int MTILES, int KSTEPS, int ACT>
__global__ __launch_bounds__(256) void k_gemm(const float* __restrict__ Xf,
                                              const float* __restrict__ Wf,
                                              const float* __restrict__ bnG,
                                              const float* __restrict__ bnB,
                                              float* __restrict__ Y,
                                              const float* __restrict__ cexp,
                                              const float* __restrict__ sep,
                                              float* __restrict__ outp) {
  constexpr int K = KSTEPS * 32;
  constexpr int M = MTILES * 16;
  constexpr int TILE = MTILES * 16 * 32;       // bf16 elems per K-step tile
  __shared__ __bf16 wlds[2][TILE];

  int lane = threadIdx.x & 31;
  int wave = threadIdx.x >> 5;
  int n0 = blockIdx.x * 128 + wave * 16;
  int b = blockIdx.y;
  const float* Xb = Xf + (size_t)b * K * HWSZ;
  int col = n0 + (lane & 15);
  int halfsel = lane >> 4;

  // stage one MTILES*16 x 32 weight tile (K-step ks) into wlds[buf], bf16
  auto stage = [&](int ks, int buf) {
    int k0 = ks * 32;
#pragma unroll
    for (int idx = (int)threadIdx.x * 4; idx < TILE; idx += 1024) {
      int m = idx >> 5, kk = idx & 31;
      float4 w4 = *(const float4*)(Wf + m * K + k0 + kk);
      v4bf p;
      p[0] = (__bf16)w4.x; p[1] = (__bf16)w4.y;
      p[2] = (__bf16)w4.z; p[3] = (__bf16)w4.w;
      *(v4bf*)&wlds[buf][idx] = p;
    }
  };

  v8f zero = {};
  v8f acc[MTILES];
#pragma unroll
  for (int mt = 0; mt < MTILES; mt++) acc[mt] = zero;

  stage(0, 0);
  __syncthreads();

  for (int ks = 0; ks < KSTEPS; ks++) {
    int buf = ks & 1;
    int k0 = ks * 32;
    if (ks + 1 < KSTEPS) stage(ks + 1, buf ^ 1);   // pipeline next tile

    // B fragment: 16 bf16 per lane (column of X), contiguous K
    v16bf bf;
    const float* xcol = Xb + (size_t)(k0 + halfsel * 16) * HWSZ + col;
#pragma unroll
    for (int e = 0; e < 16; e++) bf[e] = (__bf16)xcol[(size_t)e * HWSZ];

    if (ks + 1 < KSTEPS)
      __builtin_prefetch(Xb + (size_t)(k0 + 32 + halfsel * 16) * HWSZ + col, 0, 1);

#pragma unroll
    for (int mt = 0; mt < MTILES; mt++) {
      v16bf af;
      int mrow = mt * 16 + (lane & 15);
      int khalf = halfsel * 16;
#pragma unroll
      for (int e = 0; e < 16; e++) {
        int kk = e + khalf;
        af[e] = wlds[buf][mrow * 32 + kk];
      }
      acc[mt] = __builtin_amdgcn_wmma_f32_16x16x32_bf16(
          false, af, false, bf, (short)0, acc[mt], false, false);
    }
    __syncthreads();   // reads of wlds[buf] done; next iter may overwrite it
  }

  // Epilogue: C/D layout — VGPR r: lanes 0-15 M=r, lanes 16-31 M=r+8
  const float bnrs = rsqrtf(1.0f + 1e-5f);
#pragma unroll
  for (int mt = 0; mt < MTILES; mt++) {
#pragma unroll
    for (int r = 0; r < 8; r++) {
      int chn = mt * 16 + r + halfsel * 8;
      float v = acc[mt][r];
      v = v * (bnG[chn] * bnrs) + bnB[chn];
      if (ACT == 1) v = fmaxf(v, 0.f);
      if (ACT == 2) {
        float fw = sigmoidf(v);
        size_t idx = ((size_t)b * CHN + chn) * HWSZ + col;
        outp[idx] = cexp[idx] * fw + sep[idx] * (1.f - fw);
      } else {
        Y[((size_t)b * M + chn) * HWSZ + col] = v;
      }
    }
  }
}

// ---------------------------------------------------------------------------
extern "C" void kernel_launch(void* const* d_in, const int* in_sizes, int n_in,
                              void* d_out, int out_size, void* d_ws, size_t ws_size,
                              hipStream_t stream) {
  (void)in_sizes; (void)n_in; (void)out_size; (void)ws_size;
  const float* x          = (const float*)d_in[0];
  const float* ci_w1      = (const float*)d_in[1];
  const float* ci_w2      = (const float*)d_in[2];
  const float* ci_conv_w  = (const float*)d_in[3];
  const float* ema_conv1w = (const float*)d_in[4];
  const float* ema_conv3w = (const float*)d_in[5];
  const float* gn_g       = (const float*)d_in[6];
  const float* gn_b       = (const float*)d_in[7];
  const float* cs_conv1w  = (const float*)d_in[8];
  const float* cs_conv3w  = (const float*)d_in[9];
  const float* cs_att_w   = (const float*)d_in[10];
  const float* cs_bn_g    = (const float*)d_in[11];
  const float* cs_bn_b    = (const float*)d_in[12];
  const float* cs_cross_w = (const float*)d_in[13];
  const float* ff_w1      = (const float*)d_in[14];
  const float* ff_bn1_g   = (const float*)d_in[15];
  const float* ff_bn1_b   = (const float*)d_in[16];
  const float* ff_w2      = (const float*)d_in[17];
  const float* ff_bn2_g   = (const float*)d_in[18];
  const float* ff_bn2_b   = (const float*)d_in[19];
  const float* fin_w      = (const float*)d_in[20];
  const float* fin_bn_g   = (const float*)d_in[21];
  const float* fin_bn_b   = (const float*)d_in[22];
  float* out = (float*)d_out;

  // Workspace carve-up (fp32). Needs ~424 MB scratch.
  float* ws = (float*)d_ws;
  size_t o = 0;
  const size_t NBC = (size_t)BATCH * CHN * HWSZ;
  float* cex = ws + o; o += NBC;
  float* x1  = ws + o; o += NBC;
  float* x2  = ws + o; o += NBC;
  float* x1e = ws + o; o += NBC;
  float* x2e = ws + o; o += NBC;
  float* se  = ws + o; o += NBC;
  float* ff1 = ws + o; o += (size_t)BATCH * 64 * HWSZ;
  float* svec = ws + o; o += BATCH * CHN;
  float* catt = ws + o; o += BATCH * CHN;
  float* mh  = ws + o; o += (size_t)BG * GC * 64;
  float* mw  = ws + o; o += (size_t)BG * GC * 64;
  float* xhs = ws + o; o += (size_t)BG * GC * 64;
  float* xws = ws + o; o += (size_t)BG * GC * 64;
  float* m1  = ws + o; o += BG * GC;
  float* m2  = ws + o; o += BG * GC;
  float* ff2 = x1;  // x1 is dead after k_cross produces x1e

  k_stats <<<dim3(CHN, BATCH),   256, 0, stream>>>(x, svec);
  k_catt  <<<dim3(BATCH),        256, 0, stream>>>(svec, ci_w1, ci_w2, catt);
  k_cex   <<<dim3(16, NG, BATCH),256, 0, stream>>>(x, catt, ci_conv_w, cex);
  k_rowcol<<<dim3(GC, BG),        64, 0, stream>>>(cex, mh, mw);
  k_ema1  <<<dim3(BG),           128, 0, stream>>>(mh, mw, ema_conv1w, xhs, xws);
  k_x1gn  <<<dim3(GC, BG),       256, 0, stream>>>(cex, xhs, xws, gn_g, gn_b, x1);
  k_x2conv<<<dim3(GC, BG),       256, 0, stream>>>(cex, ema_conv3w, x2);
  k_cross <<<dim3(4, BG),        256, 0, stream>>>(x1, x1e, cs_conv1w, cs_conv3w,
                                                   cs_att_w, cs_bn_g, cs_bn_b, cs_cross_w);
  k_cross <<<dim3(4, BG),        256, 0, stream>>>(x2, x2e, cs_conv1w, cs_conv3w,
                                                   cs_att_w, cs_bn_g, cs_bn_b, cs_cross_w);
  k_pool  <<<dim3(GC, BG),       256, 0, stream>>>(x1e, x2e, m1, m2);
  k_se    <<<dim3(BG),           256, 0, stream>>>(m1, m2, x1e, x2e, cex, se);

  // FF tail: three fused BN GEMMs on the matrix units (bf16 in, fp32 acc).
  k_gemm<4, 8, 1> <<<dim3(32, BATCH), 256, 0, stream>>>(
      se, ff_w1, ff_bn1_g, ff_bn1_b, ff1, nullptr, nullptr, nullptr);       // 256->64, relu
  k_gemm<16, 2, 0><<<dim3(32, BATCH), 256, 0, stream>>>(
      ff1, ff_w2, ff_bn2_g, ff_bn2_b, ff2, nullptr, nullptr, nullptr);      // 64->256
  k_gemm<16, 8, 2><<<dim3(32, BATCH), 256, 0, stream>>>(
      ff2, fin_w, fin_bn_g, fin_bn_b, nullptr, cex, se, out);               // 256->256 + blend
}
